// deepCAT_6803228197702
// MI455X (gfx1250) — matrix operand order repaired and checked by reading, not compiled, not run on hardware
//
#include <hip/hip_runtime.h>
#include <hip/hip_bf16.h>

typedef __attribute__((ext_vector_type(16))) _Float16 v16h;
typedef __attribute__((ext_vector_type(2)))  _Float16 v2h;
typedef __attribute__((ext_vector_type(8)))  float    v8f;
typedef __attribute__((ext_vector_type(2)))  float    v2f;
typedef __attribute__((ext_vector_type(2)))  int      v2i;

#define GN    8192
#define IN_F  256
#define OUT_F 128
#define LEAK  0.2f

// ---------------------------------------------------------------------------
// Kernel 1: h = X @ W via v_wmma_f32_16x16x32_f16, stored TRANSPOSED as f16
// (hT[f][c], 128 x 8192) so the attention kernel can load WMMA B-fragments as
// contiguous 32-byte rows.  Fused si = h@a_i, sj = h@a_j.
// One wave per 16-row block; W staged transposed in LDS as f16.
// ---------------------------------------------------------------------------
__global__ __launch_bounds__(128) void gat_h_si_sj(
    const float* __restrict__ X, const float* __restrict__ W,
    const float* __restrict__ ai, const float* __restrict__ aj,
    _Float16* __restrict__ hT, float* __restrict__ si, float* __restrict__ sj)
{
    constexpr int WP = IN_F + 2;              // 258 halves: bank stride 129 == 1 mod 64
    __shared__ _Float16 Wt[OUT_F * WP];       // [f][k] transposed, ~64.5 KB

    const int tid = threadIdx.x;
    // Stage W (256x128 f32, row-major) transposed into LDS as f16.
    for (int idx = tid; idx < IN_F * OUT_F; idx += 128) {
        const int k = idx >> 7;               // 0..255
        const int f = idx & 127;              // 0..127
        Wt[f * WP + k] = (_Float16)W[idx];    // coalesced global, conflict-free LDS
    }
    __syncthreads();

    const int wave   = tid >> 5;
    const int lane   = tid & 31;
    const int r16    = lane & 15;
    const int hi     = lane >> 4;
    const int rowblk = blockIdx.x * 4 + wave; // 0..511
    const int row0   = rowblk * 16;

    v8f acc[8] = {};                          // 8 n-blocks of 16x16 f32

    const float* xrow = X + (size_t)(row0 + r16) * IN_F;

    for (int kt = 0; kt < IN_F; kt += 32) {
        // A fragment: X row r16, f16; element e -> K = e + 16*hi
        v16h a;
        #pragma unroll
        for (int v = 0; v < 8; ++v) {
            const int k0 = kt + 2 * v + 16 * hi;
            const v2f x = *(const v2f*)(xrow + k0);   // 8B global load
            a[2 * v]     = (_Float16)x[0];
            a[2 * v + 1] = (_Float16)x[1];
        }
        #pragma unroll
        for (int nb = 0; nb < 8; ++nb) {
            // B fragment: element e -> K = e + 16*hi, N = r16
            const _Float16* wp = &Wt[(nb * 16 + r16) * WP + kt + 16 * hi];
            v16h b;
            #pragma unroll
            for (int v = 0; v < 8; ++v) {
                const v2h p = *(const v2h*)(wp + 2 * v);  // one ds_load_b32
                b[2 * v]     = p[0];
                b[2 * v + 1] = p[1];
            }
            acc[nb] = __builtin_amdgcn_wmma_f32_16x16x32_f16(
                false, a, false, b, (short)0, acc[nb], false, false);
        }
    }

    // Store h TRANSPOSED as f16: hT[f][c] with row pitch GN.
    #pragma unroll
    for (int nb = 0; nb < 8; ++nb) {
        const size_t frow = (size_t)(nb * 16 + r16) * GN;
        #pragma unroll
        for (int d = 0; d < 8; ++d) {
            const int m = row0 + d + 8 * hi;
            hT[frow + m] = (_Float16)acc[nb][d];
        }
    }

    // si / sj: dot each row of h with a_i / a_j (reduce over 128 features).
    float aiv[8], ajv[8];
    #pragma unroll
    for (int nb = 0; nb < 8; ++nb) {
        aiv[nb] = ai[nb * 16 + r16];
        ajv[nb] = aj[nb * 16 + r16];
    }
    #pragma unroll
    for (int d = 0; d < 8; ++d) {
        float pi = 0.f, pj = 0.f;
        #pragma unroll
        for (int nb = 0; nb < 8; ++nb) {
            pi += acc[nb][d] * aiv[nb];
            pj += acc[nb][d] * ajv[nb];
        }
        #pragma unroll
        for (int off = 1; off < 16; off <<= 1) {   // reduce within 16-lane half
            pi += __shfl_xor(pi, off, 32);
            pj += __shfl_xor(pj, off, 32);
        }
        if (r16 == 0) {
            si[row0 + d + 8 * hi] = pi;
            sj[row0 + d + 8 * hi] = pj;
        }
    }
}

// ---------------------------------------------------------------------------
// Kernel 2: single-pass masked softmax + attention@h + per-row affine.
// One wave owns 16 rows; loops over 32-column tiles of the 8192-wide row.
// No LDS, no barriers: B-fragments are contiguous 32B rows of hT (L2/L0
// resident, 2 MB total); adj is the sole HBM stream (268 MB, read once).
// No max-subtraction needed: masked entries contribute exp -> 0 exactly,
// live logits are O(1), so p = adj ? exp(lrelu(si+sj)) : 0, l = sum p.
// ---------------------------------------------------------------------------
__global__ __launch_bounds__(256) void gat_attn(
    const int* __restrict__ adj, const _Float16* __restrict__ hT,
    const float* __restrict__ si, const float* __restrict__ sj,
    const float* __restrict__ gamma, const float* __restrict__ beta,
    float* __restrict__ out)
{
    const int tid    = threadIdx.x;
    const int wave   = tid >> 5;
    const int lane   = tid & 31;
    const int r16    = lane & 15;
    const int hi     = lane >> 4;
    const int rowblk = blockIdx.x * 8 + wave; // 0..511
    const int row0   = rowblk * 16;

    const float si_r   = si[row0 + r16];
    const int*  adjrow = adj + (size_t)(row0 + r16) * GN;
    // Per-lane base of the hT row this lane feeds into B-fragments:
    // N = r16, K-half selected by hi -> 16 contiguous halves per n-block.
    const _Float16* hrow = hT + (size_t)r16 * GN + 16 * hi;

    v8f   acc[8] = {};
    float lsum   = 0.f;

    for (int jt = 0; jt < GN; jt += 32) {
        // Prefetch next adj tile for this row while we compute this one.
        if (jt + 32 < GN)
            __builtin_prefetch(adjrow + jt + 32, 0, 0);

        // A fragment = P tile (masked exp of logits), f16.
        // Element e -> column K = e + 16*hi.
        v16h a;
        #pragma unroll
        for (int v = 0; v < 8; ++v) {
            const int K0 = 2 * v + 16 * hi;
            const v2i ad = *(const v2i*)(adjrow + jt + K0);   // 8B global load
            const v2f sv = *(const v2f*)(sj + jt + K0);       // 8B, L0-cached
            float e0 = si_r + sv[0]; e0 = (e0 > 0.f) ? e0 : LEAK * e0;
            float e1 = si_r + sv[1]; e1 = (e1 > 0.f) ? e1 : LEAK * e1;
            const float p0 = (ad[0] > 0) ? __expf(e0) : 0.f;
            const float p1 = (ad[1] > 0) ? __expf(e1) : 0.f;
            lsum += p0 + p1;
            a[2 * v]     = (_Float16)p0;
            a[2 * v + 1] = (_Float16)p1;
        }
        #pragma unroll
        for (int nb = 0; nb < 8; ++nb) {
            // B fragment: 16 contiguous halves of hT row nb*16+r16 -> 2x b128.
            const v16h b = *(const v16h*)(hrow + (size_t)nb * 16 * GN + jt);
            acc[nb] = __builtin_amdgcn_wmma_f32_16x16x32_f16(
                false, a, false, b, (short)0, acc[nb], false, false);
        }
    }

    // Full row sum: this lane's half + mirror half.
    const float lr = lsum + __shfl_xor(lsum, 16, 32);  // lane r holds l[row r16]

    #pragma unroll
    for (int d = 0; d < 8; ++d) {
        const int   m   = row0 + d + 8 * hi;
        const float ld  = __shfl(lr, d + 8 * hi, 32);  // l for row d+8*hi
        const float g   = 1.0f + gamma[m];
        const float bt  = beta[m];
        const float inv = 1.0f / ld;
        #pragma unroll
        for (int nb = 0; nb < 8; ++nb)
            out[(size_t)m * OUT_F + nb * 16 + r16] = g * acc[nb][d] * inv + bt;
    }
}

// ---------------------------------------------------------------------------
extern "C" void kernel_launch(void* const* d_in, const int* in_sizes, int n_in,
                              void* d_out, int out_size, void* d_ws, size_t ws_size,
                              hipStream_t stream) {
    (void)in_sizes; (void)n_in; (void)out_size; (void)ws_size;
    const float* X     = (const float*)d_in[0];
    const int*   adj   = (const int*)  d_in[1];
    const float* gamma = (const float*)d_in[2];
    const float* beta  = (const float*)d_in[3];
    const float* W     = (const float*)d_in[4];
    const float* ai    = (const float*)d_in[5];
    const float* aj    = (const float*)d_in[6];
    float*       out   = (float*)d_out;

    char* ws = (char*)d_ws;
    _Float16* hT = (_Float16*)ws;                                  // 2 MB, [f][c]
    float*    si = (float*)(ws + (size_t)GN * OUT_F * sizeof(_Float16));
    float*    sj = si + GN;                                        // +32KB each

    // 512 row-blocks of 16 rows.
    gat_h_si_sj<<<128, 128, 0, stream>>>(X, W, ai, aj, hT, si, sj);
    gat_attn  <<< 64, 256, 0, stream>>>(adj, hT, si, sj, gamma, beta, out);
}
